// bBetaLSTMCell_24472723652768
// MI455X (gfx1250) — matrix operand order repaired and checked by reading, not compiled, not run on hardware
//
#include <hip/hip_runtime.h>
#include <hip/hip_bf16.h>

typedef __attribute__((ext_vector_type(16))) __bf16 v16bf;
typedef __attribute__((ext_vector_type(8)))  __bf16 v8bf;
typedef __attribute__((ext_vector_type(8)))  float  v8f;

#define B_DIM 65536
#define I_DIM 512
#define H_DIM 256
#define K_TOT 768           // I + H
#define NG    1280          // 5*H gate columns
#define MT    32            // rows per workgroup
#define LDA   776           // padded bf16 row stride of staged A panel
#define BH    (B_DIM * H_DIM)
#define EPSV  0.001f

static __device__ __forceinline__ __bf16 f2bf(float f) {
  unsigned u = __builtin_bit_cast(unsigned, f);
  unsigned r = (u + 0x7FFFu + ((u >> 16) & 1u)) >> 16;   // round-to-nearest-even
  unsigned short s = (unsigned short)r;
  return __builtin_bit_cast(__bf16, s);
}

static __device__ __forceinline__ v8f wmma_bf16(v16bf a, v16bf b, v8f c) {
  // D = A(16x32) * B(32x16) + C, f32 accumulate
  return __builtin_amdgcn_wmma_f32_16x16x32_bf16(false, a, false, b, (short)0, c,
                                                 false, false);
}

// ---------------- counter-based RNG + Gamma sampler (Marsaglia-Tsang) -------
static __device__ __forceinline__ unsigned pcg(unsigned v) {
  unsigned st = v * 747796405u + 2891336453u;
  unsigned w  = ((st >> ((st >> 28) + 4u)) ^ st) * 277803737u;
  return (w >> 22) ^ w;
}
static __device__ __forceinline__ float u01(unsigned &s) {
  s = pcg(s);
  return (float)(s >> 8) * (1.0f / 16777216.0f) + (0.5f / 16777216.0f);
}
static __device__ __forceinline__ float gamma_sample(float alpha, unsigned seed) {
  unsigned s = seed * 0x9E3779B9u + 0x85EBCA6Bu;
  float boost = 1.0f, a = alpha;
  if (a < 1.0f) {                       // boost trick for alpha < 1
    float u = u01(s);
    boost = __expf(__logf(u) / alpha);  // u^(1/alpha)
    a += 1.0f;
  }
  float d = a - (1.0f / 3.0f);
  float c = rsqrtf(9.0f * d);
  float res = d;
  #pragma unroll 1
  for (int it = 0; it < 48; ++it) {
    float u1 = u01(s), u2 = u01(s);
    float x = sqrtf(-2.0f * __logf(u1)) * __cosf(6.28318530718f * u2); // normal
    float t = 1.0f + c * x;
    if (t <= 0.0f) continue;
    float v = t * t * t;
    float u = u01(s);
    if (__logf(u) < 0.5f * x * x + d * (1.0f - v + __logf(v))) { res = d * v; break; }
  }
  return res * boost;
}
static __device__ __forceinline__ float softplusf(float x) {
  return (x > 20.0f) ? x : __logf(1.0f + __expf(x));
}

// ---------------- kernel 1: pack [W_ih | W_hh] -> bf16 (1280 x 768) ---------
__global__ void pack_weights(const float* __restrict__ wih,
                             const float* __restrict__ whh,
                             __bf16* __restrict__ Wb) {
  int i = blockIdx.x * blockDim.x + threadIdx.x;
  if (i >= NG * K_TOT) return;
  int n = i / K_TOT, k = i - n * K_TOT;
  float v = (k < I_DIM) ? wih[n * I_DIM + k] : whh[n * H_DIM + (k - I_DIM)];
  Wb[i] = f2bf(v);
}

// ---------------- kernel 2: fused GEMM (WMMA bf16) + Beta-LSTM epilogue ----
__global__ __launch_bounds__(512, 1)
void beta_lstm_fused(const float* __restrict__ input_,
                     const float* __restrict__ h0,
                     const float* __restrict__ c0,
                     const float* __restrict__ bias,
                     const __bf16* __restrict__ Wb,
                     float* __restrict__ out) {
  extern __shared__ char smem[];
  __bf16* As = (__bf16*)smem;                              // MT x LDA bf16
  float*  Gs = (float*)(smem + MT * LDA * sizeof(__bf16)); // MT x NG f32

  const int tid  = threadIdx.x;
  const int row0 = blockIdx.x * MT;

  // ---- stage A panel: 32 rows x 768 cols, f32 -> bf16 into LDS ----
  for (int i = tid; i < MT * K_TOT; i += 512) {
    int r = i / K_TOT, k = i - r * K_TOT;
    float v = (k < I_DIM) ? input_[(row0 + r) * I_DIM + k]
                          : h0[(row0 + r) * H_DIM + (k - I_DIM)];
    As[r * LDA + k] = f2bf(v);
  }
  __syncthreads();

  // ---- WMMA main loop: 16 waves x (5 col-tiles x 2 row-tiles) ----
  const int wave   = tid >> 5;
  const int lane   = tid & 31;
  const int rlane  = lane & 15;          // A row within tile / C column
  const int sel8   = (lane >> 4) * 8;    // A K-half select
  const int ksel16 = (lane >> 4) * 16;   // B K-half select
  const int n0     = wave * 80;          // 5 tiles * 16 columns per wave

  v8f acc[2][5];
  #pragma unroll
  for (int rt = 0; rt < 2; ++rt)
    #pragma unroll
    for (int t = 0; t < 5; ++t)
      acc[rt][t] = (v8f){0.f, 0.f, 0.f, 0.f, 0.f, 0.f, 0.f, 0.f};

  for (int kb = 0; kb < K_TOT; kb += 32) {
    // A fragments for both 16-row tiles (from LDS, bf16 layout per ISA 7.12.2)
    v16bf a0, a1;
    {
      const __bf16* p0 = As + rlane * LDA + kb + sel8;
      const __bf16* p1 = As + (16 + rlane) * LDA + kb + sel8;
      v8bf lo0 = *(const v8bf*)(p0), hi0 = *(const v8bf*)(p0 + 16);
      v8bf lo1 = *(const v8bf*)(p1), hi1 = *(const v8bf*)(p1 + 16);
      #pragma unroll
      for (int e = 0; e < 8; ++e) {
        a0[e] = lo0[e]; a0[8 + e] = hi0[e];
        a1[e] = lo1[e]; a1[8 + e] = hi1[e];
      }
    }
    if (kb + 32 < K_TOT)  // hint next K-step of the L2-resident weight panel
      __builtin_prefetch(Wb + (n0 + rlane) * K_TOT + kb + 32 + ksel16, 0, 3);
    #pragma unroll
    for (int t = 0; t < 5; ++t) {
      // B fragment: lane = output column, K = 8h..8h+7 and 16+8h..16+8h+7 from L2
      const __bf16* wp = Wb + (n0 + t * 16 + rlane) * K_TOT + kb + sel8;
      v8bf blo = *(const v8bf*)(wp), bhi = *(const v8bf*)(wp + 16);
      v16bf b;
      #pragma unroll
      for (int e = 0; e < 8; ++e) { b[e] = blo[e]; b[8 + e] = bhi[e]; }
      acc[0][t] = wmma_bf16(a0, b, acc[0][t]);
      acc[1][t] = wmma_bf16(a1, b, acc[1][t]);
    }
  }

  // ---- scatter accumulators into the LDS gate panel ----
  const int mbase = (lane >> 4) * 8;   // C layout: lanes 0-15 -> M=r, 16-31 -> M=8+r
  #pragma unroll
  for (int rt = 0; rt < 2; ++rt)
    #pragma unroll
    for (int t = 0; t < 5; ++t) {
      const int col = n0 + t * 16 + rlane;
      #pragma unroll
      for (int r = 0; r < 8; ++r)
        Gs[(rt * 16 + mbase + r) * NG + col] = acc[rt][t][r];
    }
  __syncthreads();

  // ---- fused epilogue: softplus + Gamma sampling + cell update ----
  const int h  = tid & 255;
  const int mg = tid >> 8;             // 0 or 1 -> rows [0,16) or [16,32)
  const float b0 = bias[h];
  const float b1 = bias[H_DIM + h];
  const float b2 = bias[2 * H_DIM + h];
  const float b3 = bias[3 * H_DIM + h];
  const float b4 = bias[4 * H_DIM + h];

  for (int j = 0; j < 16; ++j) {
    const int m   = mg * 16 + j;
    const int row = row0 + m;
    const float* gr = Gs + m * NG;
    const float g0 = gr[h]              + b0;
    const float g1 = gr[H_DIM + h]      + b1;
    const float g2 = gr[2 * H_DIM + h]  + b2;
    const float go = gr[3 * H_DIM + h]  + b3;
    const float gg = gr[4 * H_DIM + h]  + b4;

    const float al0 = softplusf(g0) + EPSV;
    const float al1 = softplusf(g1) + EPSV;
    const float al2 = softplusf(g2) + EPSV;

    const unsigned idx = (unsigned)(row * H_DIM + h);
    const float G0 = gamma_sample(al0, idx * 3u + 0u) + EPSV;
    const float G1 = gamma_sample(al1, idx * 3u + 1u) + EPSV;
    const float G2 = gamma_sample(al2, idx * 3u + 2u) + EPSV;

    const float a_i = G1, b_i = G2, a_f = G0, b_f = G2;
    const float si = a_i / (a_i + b_i);
    const float sf = a_f / (a_f + b_f);
    const float c1 = sf * c0[row * H_DIM + h] + si * tanhf(gg);
    const float h1 = (1.0f / (1.0f + __expf(-go))) * tanhf(c1);

    const int e = row * H_DIM + h;
    out[0 * BH + e] = h1;
    out[1 * BH + e] = c1;
    out[2 * BH + e] = a_i;
    out[3 * BH + e] = b_i;
    out[4 * BH + e] = a_f;
    out[5 * BH + e] = b_f;
  }
}

extern "C" void kernel_launch(void* const* d_in, const int* in_sizes, int n_in,
                              void* d_out, int out_size, void* d_ws, size_t ws_size,
                              hipStream_t stream) {
  const float* input_ = (const float*)d_in[0];
  const float* h0     = (const float*)d_in[1];
  const float* c0     = (const float*)d_in[2];
  const float* w_ih   = (const float*)d_in[3];
  const float* w_hh   = (const float*)d_in[4];
  const float* bias   = (const float*)d_in[5];
  float* out = (float*)d_out;
  __bf16* Wb = (__bf16*)d_ws;   // 1280*768 bf16 = 1.92 MB (L2-resident)

  const int wn = NG * K_TOT;
  pack_weights<<<(wn + 255) / 256, 256, 0, stream>>>(w_ih, w_hh, Wb);

  const size_t smem_bytes = (size_t)MT * LDA * sizeof(__bf16)   // 49,664 B
                          + (size_t)MT * NG  * sizeof(float);   // 163,840 B
  beta_lstm_fused<<<B_DIM / MT, 512, smem_bytes, stream>>>(
      input_, h0, c0, bias, Wb, out);
}
